// MultiHeadSelfAttention_69535520522287
// MI455X (gfx1250) — matrix operand order repaired and checked by reading, not compiled, not run on hardware
//
#include <hip/hip_runtime.h>
#include <hip/hip_bf16.h>
#include <stdint.h>

// ---------------------------------------------------------------------------
// MHA forward for MI455X (gfx1250), wave32 + WMMA bf16 (K=32) matrix cores.
// Pipeline:
//   1) gemm_bias<fp32A,split> x3 : Q/K/V projections -> bf16 [B,H,S,D]
//   2) attn_kernel              : causal flash-attention -> bf16 [B,S,E]
//                                 K/V double-buffered via async global->LDS
//                                 DMA (ASYNCcnt partial waits), DPP softmax
//   3) gemm_bias<bf16A,dense>   : output projection -> fp32 d_out
// ---------------------------------------------------------------------------

typedef __attribute__((ext_vector_type(16))) __bf16 v16bf;
typedef __attribute__((ext_vector_type(2)))  __bf16 v2bf;
typedef __attribute__((ext_vector_type(8)))  float  v8f;

constexpr int Bc = 2, Sc = 2048, Ec = 1024, Hc = 16, Dc = 64;
constexpr int Mtot = Bc * Sc;   // 4096 rows for all projections

// float -> bf16, round-to-nearest-even
__device__ inline __bf16 f2bf(float f) {
    unsigned u = __builtin_bit_cast(unsigned, f);
    unsigned r = u + 0x7FFFu + ((u >> 16) & 1u);
    unsigned short h = (unsigned short)(r >> 16);
    return __builtin_bit_cast(__bf16, h);
}

// packed pair conversion (single v_cvt_pk_bf16_f32 when the builtin exists)
__device__ inline v2bf pk_bf16(float lo, float hi) {
#if __has_builtin(__builtin_amdgcn_cvt_pk_bf16_f32)
    return __builtin_amdgcn_cvt_pk_bf16_f32(lo, hi);
#else
    v2bf r; r[0] = f2bf(lo); r[1] = f2bf(hi); return r;
#endif
}

__device__ inline v8f vzero8() {
    v8f z;
#pragma unroll
    for (int i = 0; i < 8; ++i) z[i] = 0.0f;
    return z;
}

// Low 32 bits of a flat pointer to __shared__ == wave-relative LDS byte addr.
__device__ inline unsigned lds_off(const void* p) {
    return (unsigned)(uintptr_t)p;
}

// Async DMA: 16 bytes global -> LDS, tracked by ASYNCcnt (CDNA5 path).
// GVS addressing: mem = SGPR64 + VGPR32 byte offset.
__device__ inline void async_ld_b128(unsigned lds_addr, unsigned gbyte_off,
                                     const void* sbase) {
    asm volatile("global_load_async_to_lds_b128 %0, %1, %2"
                 :
                 : "v"(lds_addr), "v"(gbyte_off),
                   "s"((unsigned long long)(uintptr_t)sbase)
                 : "memory");
}
template <int N>
__device__ inline void wait_async() {
    asm volatile("s_wait_asynccnt %0" :: "i"(N) : "memory");
}

// 16-lane rotate (DPP16 row_ror:N) — stays inside each 16-lane row on wave32.
template <int N>
__device__ inline float dpp_ror16(float x) {
    int xi = __builtin_bit_cast(int, x);
    int r  = __builtin_amdgcn_update_dpp(xi, xi, 0x120 + N, 0xF, 0xF, false);
    return __builtin_bit_cast(float, r);
}
// full 16-lane reductions via rotate-accumulate (pure VALU, no LDS round-trip)
__device__ inline float rowmax16(float v) {
    v = fmaxf(v, dpp_ror16<1>(v));
    v = fmaxf(v, dpp_ror16<2>(v));
    v = fmaxf(v, dpp_ror16<4>(v));
    v = fmaxf(v, dpp_ror16<8>(v));
    return v;
}
__device__ inline float rowsum16(float v) {
    v += dpp_ror16<1>(v);
    v += dpp_ror16<2>(v);
    v += dpp_ror16<4>(v);
    v += dpp_ror16<8>(v);
    return v;
}

// A fragment (16x32, MxK) from row-major LDS [M][K], stride ld elements.
// Lane L: row = L%16; K-halves {0..7,16..23} (L<16) / {8..15,24..31} (L>=16),
// packed 2 bf16 per VGPR (elements 2g, 2g+1 = K, K+1).
__device__ inline v16bf frag_a_ld(const __bf16* p, int ld) {
    const int lane = threadIdx.x & 31;
    const int row  = lane & 15;
    const int kh   = (lane >> 4) << 3;   // 0 or 8
    v16bf f;
#pragma unroll
    for (int g = 0; g < 8; ++g) {
        int k = ((g & 4) ? 16 : 0) + kh + ((g & 3) << 1);
        f[2 * g]     = p[row * ld + k];
        f[2 * g + 1] = p[row * ld + k + 1];
    }
    return f;
}

// B fragment (32x16, KxN) from row-major LDS [K][N], stride ld elements.
__device__ inline v16bf frag_b_ld(const __bf16* p, int ld) {
    const int lane = threadIdx.x & 31;
    const int col  = lane & 15;
    const int kh   = (lane >> 4) << 3;
    v16bf f;
#pragma unroll
    for (int g = 0; g < 8; ++g) {
        int k = ((g & 4) ? 16 : 0) + kh + ((g & 3) << 1);
        f[2 * g]     = p[k * ld + col];
        f[2 * g + 1] = p[(k + 1) * ld + col];
    }
    return f;
}

__device__ inline v8f wmma_bf16(v16bf a, v16bf b, v8f c) {
    // (neg_a, A, neg_b, B, c_mod, C, reuse_a, reuse_b)
    return __builtin_amdgcn_wmma_f32_16x16x32_bf16(false, a, false, b,
                                                   (short)0, c, false, false);
}

// ---------------------------------------------------------------------------
// Generic GEMM: C[M,N] = A[M,K] * W[K,N] + bias.  A fp32 or bf16 (row-major),
// W/bias fp32.  SPLIT_HEADS: store bf16 into [B,H,S,D]; else fp32 [M,N].
// Block tile 64x128, BK=32; 8 waves (2x4), 32x32 per wave -> 4 wmma / K-step.
// bf16-A tiles are staged with async global->LDS B128 DMA.
// ---------------------------------------------------------------------------
template <bool A_BF16, bool SPLIT_HEADS>
__global__ __launch_bounds__(256) void gemm_bias(const void* __restrict__ Ap,
                                                 const float* __restrict__ W,
                                                 const float* __restrict__ bias,
                                                 void* __restrict__ Out,
                                                 int M, int K, int N) {
    constexpr int BM = 64, BN = 128, BK = 32;
    constexpr int LDA = 40;                 // 80 B stride: 16B-aligned, banks spread
    __shared__ __bf16 As[BM * LDA];
    __shared__ __bf16 Bs[BK * BN];

    const int tid  = threadIdx.x;
    const int wave = tid >> 5;
    const int lane = tid & 31;
    const int wm   = (wave >> 2) * 32;          // 0 / 32
    const int wn   = (wave & 3) * 32;           // 0..96
    const int bm   = blockIdx.x * BM;
    const int bn   = blockIdx.y * BN;

    v8f acc[2][2];
#pragma unroll
    for (int i = 0; i < 2; ++i)
#pragma unroll
        for (int j = 0; j < 2; ++j) acc[i][j] = vzero8();

    for (int k0 = 0; k0 < K; k0 += BK) {
        __syncthreads();
        // stage A tile
        if constexpr (A_BF16) {
            // 64x32 bf16 = 4096 B = 256 x B128 chunks, one per thread
            const __bf16* A = (const __bf16*)Ap;
            int r  = tid >> 2;                  // 4 chunks per 64-B row
            int c8 = tid & 3;                   // chunk within row (8 elems)
            unsigned l = lds_off(&As[r * LDA + c8 * 8]);
            unsigned g = (unsigned)(((size_t)(bm + r) * K + k0 + c8 * 8) * 2);
            async_ld_b128(l, g, A);
        } else {
            const float* A = (const float*)Ap;
#pragma unroll
            for (int i = 0; i < (BM * BK) / 512; ++i) {     // pairs
                int e2 = tid + i * 256;
                int r = e2 >> 4, c2 = e2 & 15;
                const float* src = &A[(size_t)(bm + r) * K + k0 + c2 * 2];
                *(v2bf*)&As[r * LDA + c2 * 2] = pk_bf16(src[0], src[1]);
            }
        }
        // stage W tile fp32->bf16 (conversion, must go through VGPRs)
#pragma unroll
        for (int i = 0; i < (BK * BN) / 512; ++i) {         // pairs
            int e2 = tid + i * 256;
            int r = e2 >> 6, c2 = e2 & 63;
            const float* src = &W[(size_t)(k0 + r) * N + bn + c2 * 2];
            *(v2bf*)&Bs[r * BN + c2 * 2] = pk_bf16(src[0], src[1]);
        }
        if (k0 + BK < K)  // hint next W tile toward L2 (global_prefetch_b8)
            __builtin_prefetch(&W[(size_t)(k0 + BK) * N + bn + (tid & 127)], 0, 0);
        if constexpr (A_BF16) wait_async<0>();
        __syncthreads();

        v16bf a0 = frag_a_ld(&As[(wm + 0) * LDA], LDA);
        v16bf a1 = frag_a_ld(&As[(wm + 16) * LDA], LDA);
        v16bf b0 = frag_b_ld(&Bs[wn + 0], BN);
        v16bf b1 = frag_b_ld(&Bs[wn + 16], BN);
        acc[0][0] = wmma_bf16(a0, b0, acc[0][0]);
        acc[0][1] = wmma_bf16(a0, b1, acc[0][1]);
        acc[1][0] = wmma_bf16(a1, b0, acc[1][0]);
        acc[1][1] = wmma_bf16(a1, b1, acc[1][1]);
    }

    // C/D layout: VGPR r -> row r + 8*(lane>=16), col = lane%16
    const int mo = (lane >> 4) << 3;
    const int nn = lane & 15;
#pragma unroll
    for (int tm = 0; tm < 2; ++tm)
#pragma unroll
        for (int tn = 0; tn < 2; ++tn)
#pragma unroll
            for (int r = 0; r < 8; ++r) {
                int m = bm + wm + tm * 16 + mo + r;
                int n = bn + wn + tn * 16 + nn;
                float val = acc[tm][tn][r] + bias[n];
                if constexpr (SPLIT_HEADS) {
                    __bf16* O = (__bf16*)Out;
                    int b = m >> 11;            // /Sc
                    int s = m & (Sc - 1);
                    int h = n >> 6;             // /Dc
                    int d = n & (Dc - 1);
                    O[(((size_t)b * Hc + h) * Sc + s) * Dc + d] = f2bf(val);
                } else {
                    float* O = (float*)Out;
                    O[(size_t)m * N + n] = val;
                }
            }
}

// ---------------------------------------------------------------------------
// Causal flash attention. Block = 64 query rows of one (b,h); 128 threads,
// 4 waves x 16 rows.  K/V tiles double-buffered via async global->LDS B128
// DMA: issue tile i+1, s_wait_asynccnt 4 (in-order) => tile i landed while
// tile i+1 flies during compute.  Softmax reductions in pure VALU (DPP16).
// ---------------------------------------------------------------------------
__global__ __launch_bounds__(128) void attn_kernel(const __bf16* __restrict__ qh,
                                                   const __bf16* __restrict__ kh,
                                                   const __bf16* __restrict__ vh,
                                                   __bf16* __restrict__ attn) {
    constexpr int LDQ = 72;                 // 144 B stride: 16B-aligned, bank spread
    constexpr int LDP = 36;
    __shared__ __bf16 Qs[64 * LDQ];
    __shared__ __bf16 Ks[2][32 * LDQ];
    __shared__ __bf16 Vs[2][32 * Dc];       // frag_b walks N across lanes -> ok
    __shared__ __bf16 Ps[4 * 16 * LDP];

    const int qb   = blockIdx.x;            // query block of 64
    const int bh   = blockIdx.y;            // b*H + h
    const int b    = bh >> 4;               // /Hc
    const int h    = bh & (Hc - 1);
    const int tid  = threadIdx.x;
    const int wave = tid >> 5;
    const int lane = tid & 31;
    const size_t baseB = (size_t)bh * Sc * Dc * 2;  // bytes

    // per-thread chunk coords for 32x64 bf16 tiles (256 x 16B chunks, 2/thread)
    const int tr0 = tid >> 3, tc0 = (tid & 7) * 16;          // chunk 0
    const int tr1 = (tid + 128) >> 3, tc1 = tc0;             // chunk 1 (rows 16..31)

    // ---- Q tile 64x64 bf16 (8 KiB) via async DMA: 512 chunks, 4/thread
#pragma unroll
    for (int i = 0; i < 4; ++i) {
        int e  = tid + i * 128;
        int r  = e >> 3;                    // 8 chunks per 128-B row
        int c8 = e & 7;
        unsigned l = lds_off(&Qs[r * LDQ + c8 * 8]);
        unsigned g = (unsigned)((size_t)(qb * 64 + r) * Dc * 2 + c8 * 16);
        async_ld_b128(l, g, (const char*)qh + baseB);
    }
    // ---- prologue: K/V tile for kv=0 into buffer 0 (4 ops/thread)
    {
        unsigned g0 = (unsigned)((size_t)tr0 * Dc * 2 + tc0);
        unsigned g1 = (unsigned)((size_t)tr1 * Dc * 2 + tc1);
        async_ld_b128(lds_off((char*)&Ks[0][0] + tr0 * LDQ * 2 + tc0), g0,
                      (const char*)kh + baseB);
        async_ld_b128(lds_off((char*)&Vs[0][0] + tr0 * Dc * 2 + tc0), g0,
                      (const char*)vh + baseB);
        async_ld_b128(lds_off((char*)&Ks[0][0] + tr1 * LDQ * 2 + tc1), g1,
                      (const char*)kh + baseB);
        async_ld_b128(lds_off((char*)&Vs[0][0] + tr1 * Dc * 2 + tc1), g1,
                      (const char*)vh + baseB);
    }
    wait_async<0>();
    __syncthreads();
    const v16bf aq0 = frag_a_ld(&Qs[wave * 16 * LDQ], LDQ);       // d 0..31
    const v16bf aq1 = frag_a_ld(&Qs[wave * 16 * LDQ + 32], LDQ);  // d 32..63

    float mi[8], li[8];
    v8f   O[4];
#pragma unroll
    for (int r = 0; r < 8; ++r) { mi[r] = -1e30f; li[r] = 0.0f; }
#pragma unroll
    for (int t = 0; t < 4; ++t) O[t] = vzero8();

    const int mo = (lane >> 4) << 3;
    const int nn = lane & 15;
    const int kv_end = (qb + 1) * 64;

    for (int kv = 0; kv < kv_end; kv += 32) {
        const int cur  = (kv >> 5) & 1;
        const int nxt  = cur ^ 1;
        const bool more = (kv + 32) < kv_end;

        __syncthreads();                 // WAR: iter i-1 readers done with nxt
        if (more) {                      // prefetch tile i+1 into nxt
            unsigned g0 = (unsigned)((size_t)(kv + 32 + tr0) * Dc * 2 + tc0);
            unsigned g1 = (unsigned)((size_t)(kv + 32 + tr1) * Dc * 2 + tc1);
            async_ld_b128(lds_off((char*)&Ks[nxt][0] + tr0 * LDQ * 2 + tc0), g0,
                          (const char*)kh + baseB);
            async_ld_b128(lds_off((char*)&Vs[nxt][0] + tr0 * Dc * 2 + tc0), g0,
                          (const char*)vh + baseB);
            async_ld_b128(lds_off((char*)&Ks[nxt][0] + (tr1 - 16 + 16) * LDQ * 2 + tc1), g1,
                          (const char*)kh + baseB);
            async_ld_b128(lds_off((char*)&Vs[nxt][0] + (tr1 - 16 + 16) * Dc * 2 + tc1), g1,
                          (const char*)vh + baseB);
            wait_async<4>();             // in-order: tile i landed, i+1 flies
        } else {
            wait_async<0>();
        }
        __syncthreads();                 // tile i visible from all waves

        const __bf16* Kc = &Ks[cur][0];
        const __bf16* Vc = &Vs[cur][0];

        // logits S = Q * K^T : keys as B-columns via A-pattern load from [key][d]
        v8f s0 = vzero8(), s1 = vzero8();
        s0 = wmma_bf16(aq0, frag_a_ld(&Kc[0], LDQ), s0);
        s0 = wmma_bf16(aq1, frag_a_ld(&Kc[32], LDQ), s0);
        s1 = wmma_bf16(aq0, frag_a_ld(&Kc[16 * LDQ], LDQ), s1);
        s1 = wmma_bf16(aq1, frag_a_ld(&Kc[16 * LDQ + 32], LDQ), s1);

        // online softmax, per register-row r (row m = mo + r); DPP reductions
#pragma unroll
        for (int r = 0; r < 8; ++r) {
            int m_g = qb * 64 + wave * 16 + mo + r;
            float x0 = s0[r] * 0.125f;          // 1/sqrt(64)
            float x1 = s1[r] * 0.125f;
            if (kv + nn      > m_g) x0 = -1e30f;
            if (kv + 16 + nn > m_g) x1 = -1e30f;
            float nm = fmaxf(mi[r], rowmax16(fmaxf(x0, x1)));
            float p0 = __expf(x0 - nm);
            float p1 = __expf(x1 - nm);
            float rs = rowsum16(p0 + p1);
            float alpha = __expf(mi[r] - nm);
            li[r] = li[r] * alpha + rs;
            mi[r] = nm;
#pragma unroll
            for (int t = 0; t < 4; ++t) O[t][r] *= alpha;
            int m_l = mo + r;
            Ps[(wave * 16 + m_l) * LDP + nn]      = f2bf(p0);
            Ps[(wave * 16 + m_l) * LDP + 16 + nn] = f2bf(p1);
        }

        // O += P(16x32) * V(32x64)  — per-wave LDS, DS in-order within wave
        v16bf ap = frag_a_ld(&Ps[wave * 16 * LDP], LDP);
#pragma unroll
        for (int t = 0; t < 4; ++t)
            O[t] = wmma_bf16(ap, frag_b_ld(&Vc[t * 16], Dc), O[t]);
    }

    // normalize + store bf16 [B,S,E] for the output projection
#pragma unroll
    for (int t = 0; t < 4; ++t)
#pragma unroll
        for (int r = 0; r < 8; ++r) {
            int s_g = qb * 64 + wave * 16 + mo + r;
            int e   = h * Dc + t * 16 + nn;
            attn[((size_t)b * Sc + s_g) * Ec + e] = f2bf(O[t][r] / li[r]);
        }
}

// ---------------------------------------------------------------------------
extern "C" void kernel_launch(void* const* d_in, const int* in_sizes, int n_in,
                              void* d_out, int out_size, void* d_ws, size_t ws_size,
                              hipStream_t stream) {
    (void)in_sizes; (void)n_in; (void)out_size; (void)ws_size;
    const float* v  = (const float*)d_in[0];
    const float* k  = (const float*)d_in[1];
    const float* q  = (const float*)d_in[2];
    /* d_in[3] = mask: causal, applied analytically in attn_kernel */
    const float* wq = (const float*)d_in[4];
    const float* bq = (const float*)d_in[5];
    const float* wk = (const float*)d_in[6];
    const float* bk = (const float*)d_in[7];
    const float* wv = (const float*)d_in[8];
    const float* bv = (const float*)d_in[9];
    const float* wo = (const float*)d_in[10];
    const float* bo = (const float*)d_in[11];

    const size_t NTOK = (size_t)Mtot * Ec;      // 4096*1024 elements
    __bf16* qh = (__bf16*)d_ws;
    __bf16* kh = qh + NTOK;
    __bf16* vh = kh + NTOK;
    __bf16* at = vh + NTOK;

    dim3 gg(Mtot / 64, Ec / 128);               // 64 x 8
    gemm_bias<false, true><<<gg, 256, 0, stream>>>(q, wq, bq, qh, Mtot, Ec, Ec);
    gemm_bias<false, true><<<gg, 256, 0, stream>>>(k, wk, bk, kh, Mtot, Ec, Ec);
    gemm_bias<false, true><<<gg, 256, 0, stream>>>(v, wv, bv, vh, Mtot, Ec, Ec);

    attn_kernel<<<dim3(Sc / 64, Bc * Hc), 128, 0, stream>>>(qh, kh, vh, at);

    gemm_bias<true, false><<<gg, 256, 0, stream>>>(at, wo, bo, d_out, Mtot, Ec, Ec);
}
